// Experts_45775761441023
// MI455X (gfx1250) — matrix operand order repaired and checked by reading, not compiled, not run on hardware
//
#include <hip/hip_runtime.h>
#include <math.h>

typedef __bf16 bf16;
typedef __attribute__((ext_vector_type(16))) __bf16 v16bf;
typedef __attribute__((ext_vector_type(8)))  __bf16 v8bf;
typedef __attribute__((ext_vector_type(8)))  float  v8f;

#define B_    4
#define NTOK  2048
#define DIM_  512
#define NEXP  16
#define HID   512
#define ODIM  128
#define NTILE 128      // token rows per workgroup
#define HC    64       // hidden-chunk width
#define LDA   520      // padded LDS row (elements) for sX / sW1 (1040 B, 16B-aligned)
#define LDB   72       // padded LDS row (elements) for sW2 / sH  (144 B, 16B-aligned)

#define SX_BYTES  (NTILE * LDA * 2)   // 133120
#define SW1_BYTES (HC    * LDA * 2)   //  66560
#define SW2_BYTES (ODIM  * LDB * 2)   //  18432
#define SH_BYTES  (NTILE * LDB * 2)   //  18432
#define SMEM_BYTES (SX_BYTES + SW1_BYTES + SW2_BYTES + SH_BYTES)  // 236544

// A-matrix fragment (16x32 bf16), contiguous K layout (matches load_frag_b):
//   lanes 0-15 : M=lane,    K=0..15
//   lanes 16-31: M=lane-16, K=16..31
static __device__ __forceinline__ v16bf load_frag_a(const bf16* base, int ld, int k0, int lane) {
    const int m  = lane & 15;
    const int kh = (lane >> 4) << 4;           // 0 or 16
    const bf16* p = base + m * ld + k0 + kh;
    v8bf lo = *(const v8bf*)(p);               // K = kh + 0..7   (ds_load_b128)
    v8bf hi = *(const v8bf*)(p + 8);           // K = kh + 8..15  (ds_load_b128)
    return __builtin_shufflevector(lo, hi, 0,1,2,3,4,5,6,7,8,9,10,11,12,13,14,15);
}

// B-matrix fragment (32x16 bf16) from transposed [n][k] LDS staging:
//   lanes 0-15 : N=lane,    K=0..15 ; lanes 16-31: N=lane-16, K=16..31
static __device__ __forceinline__ v16bf load_frag_b(const bf16* base, int ld, int n0, int k0, int lane) {
    const int n  = lane & 15;
    const int kh = (lane >> 4) << 4;           // 0 or 16
    const bf16* p = base + (n0 + n) * ld + k0 + kh;
    v8bf lo = *(const v8bf*)(p);
    v8bf hi = *(const v8bf*)(p + 8);
    return __builtin_shufflevector(lo, hi, 0,1,2,3,4,5,6,7,8,9,10,11,12,13,14,15);
}

__global__ __launch_bounds__(256) void experts_fused_kernel(
    const float* __restrict__ x, const float* __restrict__ w1,
    const float* __restrict__ w2, float* __restrict__ out) {
    extern __shared__ __align__(16) char smem[];
    bf16* sX  = (bf16*)(smem);                                   // [NTILE][LDA]  x tile (bf16)
    bf16* sW1 = (bf16*)(smem + SX_BYTES);                        // [HC][LDA]     w1 chunk, transposed [n][k]
    bf16* sW2 = (bf16*)(smem + SX_BYTES + SW1_BYTES);            // [ODIM][LDB]   w2 chunk, transposed [o][k]
    bf16* sH  = (bf16*)(smem + SX_BYTES + SW1_BYTES + SW2_BYTES);// [NTILE][LDB]  gelu(hidden) chunk

    const int tid  = threadIdx.x;
    const int lane = tid & 31;
    const int wave = tid >> 5;                 // 8 waves
    const int wm   = wave & 3;                 // M-stripe: rows [wm*32, wm*32+32)
    const int ws   = wave >> 2;                // N-split: GEMM1 cols ws*32.., GEMM2 cols ws*64..
    const int ncol = lane & 15;                // C/D layout: N = lane%16
    const int mb   = (lane >> 4) << 3;         // C/D layout: M base = 0 or 8

    const int bid = blockIdx.x;
    const int e   = bid & 15;
    const int nt  = (bid >> 4) & 15;
    const int b   = bid >> 8;
    const int n0  = nt * NTILE;

    const float* xg  = x  + ((size_t)b * NTOK + n0) * DIM_;
    const float* w1g = w1 + (size_t)e * DIM_ * HID;
    const float* w2g = w2 + (size_t)e * HID * ODIM;

    // ---- stage x tile (128x512 f32 -> bf16), float4-coalesced ----
    #pragma unroll
    for (int it = 0; it < 64; ++it) {
        const int idx = (it * 256 + tid) * 4;          // element index, %4 == 0
        const int m = idx >> 9, k = idx & 511;
        const float4 f = *(const float4*)(xg + (size_t)m * DIM_ + k);
        bf16* d = sX + m * LDA + k;
        d[0] = (bf16)f.x; d[1] = (bf16)f.y; d[2] = (bf16)f.z; d[3] = (bf16)f.w;
    }

    const v8f vzero = {0.f,0.f,0.f,0.f,0.f,0.f,0.f,0.f};
    v8f oacc[2][4];                            // 32 rows x 64 out-cols per wave
    #pragma unroll
    for (int ti = 0; ti < 2; ++ti)
        #pragma unroll
        for (int tj = 0; tj < 4; ++tj) oacc[ti][tj] = vzero;

    for (int c = 0; c < HID / HC; ++c) {
        const int hc = c * HC;
        __syncthreads();  // prior chunk done reading sW1/sW2 (also fences sX staging on c==0)

        // ---- stage w1[:, hc:hc+64] -> sW1[n][k] (transposed, bf16) ----
        #pragma unroll
        for (int it = 0; it < 32; ++it) {
            const int idx = (it * 256 + tid) * 4;
            const int k = idx >> 6, n = idx & 63;      // n % 4 == 0
            const float4 f = *(const float4*)(w1g + (size_t)k * HID + hc + n);
            sW1[(n + 0) * LDA + k] = (bf16)f.x;
            sW1[(n + 1) * LDA + k] = (bf16)f.y;
            sW1[(n + 2) * LDA + k] = (bf16)f.z;
            sW1[(n + 3) * LDA + k] = (bf16)f.w;
        }
        // ---- stage w2[hc:hc+64, :] -> sW2[o][k] (transposed, bf16) ----
        #pragma unroll
        for (int it = 0; it < 8; ++it) {
            const int idx = (it * 256 + tid) * 4;
            const int k = idx >> 7, o = idx & 127;     // o % 4 == 0
            const float4 f = *(const float4*)(w2g + (size_t)(hc + k) * ODIM + o);
            sW2[(o + 0) * LDB + k] = (bf16)f.x;
            sW2[(o + 1) * LDB + k] = (bf16)f.y;
            sW2[(o + 2) * LDB + k] = (bf16)f.z;
            sW2[(o + 3) * LDB + k] = (bf16)f.w;
        }
        __syncthreads();

        // ---- GEMM1: 32x32 tile per wave = xTile[wm*32.., :] @ w1chunk[:, ws*32..] ----
        v8f hacc[2][2];
        #pragma unroll
        for (int ti = 0; ti < 2; ++ti)
            #pragma unroll
            for (int tj = 0; tj < 2; ++tj) hacc[ti][tj] = vzero;

        #pragma unroll
        for (int k0 = 0; k0 < DIM_; k0 += 32) {
            v16bf a0 = load_frag_a(sX + (wm * 32 +  0) * LDA, LDA, k0, lane);
            v16bf a1 = load_frag_a(sX + (wm * 32 + 16) * LDA, LDA, k0, lane);
            v16bf b0 = load_frag_b(sW1, LDA, ws * 32 +  0, k0, lane);
            v16bf b1 = load_frag_b(sW1, LDA, ws * 32 + 16, k0, lane);
            hacc[0][0] = __builtin_amdgcn_wmma_f32_16x16x32_bf16(false, a0, false, b0, (short)0, hacc[0][0], false, false);
            hacc[0][1] = __builtin_amdgcn_wmma_f32_16x16x32_bf16(false, a0, false, b1, (short)0, hacc[0][1], false, false);
            hacc[1][0] = __builtin_amdgcn_wmma_f32_16x16x32_bf16(false, a1, false, b0, (short)0, hacc[1][0], false, false);
            hacc[1][1] = __builtin_amdgcn_wmma_f32_16x16x32_bf16(false, a1, false, b1, (short)0, hacc[1][1], false, false);
        }

        // ---- exact GELU, spill wave's 32x32 tile to sH as bf16 ----
        #pragma unroll
        for (int ti = 0; ti < 2; ++ti) {
            #pragma unroll
            for (int tj = 0; tj < 2; ++tj) {
                #pragma unroll
                for (int v = 0; v < 8; ++v) {
                    float val = hacc[ti][tj][v];
                    val = 0.5f * val * (1.0f + erff(val * 0.70710678118654752f));
                    sH[(wm * 32 + ti * 16 + mb + v) * LDB + (ws * 32 + tj * 16 + ncol)] = (bf16)val;
                }
            }
        }
        __syncthreads();   // sH rows are consumed cross-wave below

        // ---- GEMM2: oacc[32 x 64] += gelu_chunk[wm*32.., :64] @ w2chunk[:, ws*64..] ----
        #pragma unroll
        for (int kk = 0; kk < HC; kk += 32) {
            v16bf a0 = load_frag_a(sH + (wm * 32 +  0) * LDB, LDB, kk, lane);
            v16bf a1 = load_frag_a(sH + (wm * 32 + 16) * LDB, LDB, kk, lane);
            #pragma unroll
            for (int tj = 0; tj < 4; ++tj) {
                v16bf bm = load_frag_b(sW2, LDB, ws * 64 + tj * 16, kk, lane);
                oacc[0][tj] = __builtin_amdgcn_wmma_f32_16x16x32_bf16(false, a0, false, bm, (short)0, oacc[0][tj], false, false);
                oacc[1][tj] = __builtin_amdgcn_wmma_f32_16x16x32_bf16(false, a1, false, bm, (short)0, oacc[1][tj], false, false);
            }
        }
    }

    // ---- store: out[b][n][o][e], n = n0 + wm*32 + ti*16 + mb + v, o = ws*64 + tj*16 + ncol ----
    #pragma unroll
    for (int ti = 0; ti < 2; ++ti) {
        const size_t rowBase = (((size_t)b * NTOK + n0 + wm * 32 + ti * 16 + mb) * ODIM) * NEXP + e;
        #pragma unroll
        for (int tj = 0; tj < 4; ++tj) {
            #pragma unroll
            for (int v = 0; v < 8; ++v) {
                out[rowBase + (size_t)v * ODIM * NEXP
                            + (size_t)(ws * 64 + tj * 16 + ncol) * NEXP] = oacc[ti][tj][v];
            }
        }
    }
}

extern "C" void kernel_launch(void* const* d_in, const int* in_sizes, int n_in,
                              void* d_out, int out_size, void* d_ws, size_t ws_size,
                              hipStream_t stream) {
    const float* x   = (const float*)d_in[0];
    const float* w1  = (const float*)d_in[1];
    const float* w2  = (const float*)d_in[2];
    float*       out = (float*)d_out;
    (void)in_sizes; (void)n_in; (void)out_size; (void)d_ws; (void)ws_size;

    dim3 grid(B_ * (NTOK / NTILE) * NEXP);   // 4 * 16 * 16 = 1024 workgroups
    dim3 block(256);                          // 8 wave32 per WG
    hipLaunchKernelGGL(experts_fused_kernel, grid, block, SMEM_BYTES, stream,
                       x, w1, w2, out);
}
